// RNNdiff_71536975282220
// MI455X (gfx1250) — matrix operand order repaired and checked by reading, not compiled, not run on hardware
//
#include <hip/hip_runtime.h>
#include <hip/hip_bf16.h>
#include <math.h>

// Problem constants (from reference)
#define B_   128
#define V_   50
#define KC_  40
#define D_   256
#define H_   256
#define NH_  8
#define HD_  32

typedef __attribute__((ext_vector_type(16))) _Float16 v16h;
typedef __attribute__((ext_vector_type(8)))  _Float16 v8h;
typedef __attribute__((ext_vector_type(8)))  float    v8f;

// ---------------------------------------------------------------------------
// f16 WMMA GEMM:  C[M,N] = A[M,K] @ Bw[N,K]^T (+bias) (opt relu)
// Register-blocked: each wave computes a 16x64 tile (4 accumulators) so one
// A fragment feeds 4 WMMAs.  Block = 4 waves -> 64x64 tile per block.
// Loop body is split into a load phase (A + 4 B fragments, clause-grouped)
// and a compute phase (4 back-to-back WMMAs) so the scheduler can overlap
// the next iteration's loads with the current WMMAs (partial loadcnt waits).
// Fragment layouts (ISA 7.12.2 / 7.12.4):
//   A (16x32): lane m=(L&15), K-groups {koff..+7, koff+16..+23}, koff=(L>>4)*8
//   B (32x16): lane n=(L&15), K-groups {kb2..+7, kb2+16..+23}, kb2=(L>>4)*8
//   C/D (16x16 f32): lane L: n=(L&15), rows m=(L>>4)*8+g in acc[g]
// Requires M%64==0, N%64==0, K%32==0 (true for all call sites).
// ---------------------------------------------------------------------------
__global__ void wmma_gemm_f16(const _Float16* __restrict__ A, long lda,
                              const _Float16* __restrict__ Bw,   // [N,K] row-major
                              const float* __restrict__ bias,    // [N] or null
                              float* __restrict__ Cf, long ldcf, // f32 out or null
                              _Float16* __restrict__ Ch, long ldch, // f16 out or null
                              int K, int do_relu)
{
    const int lane   = threadIdx.x & 31;
    const int wave   = threadIdx.x >> 5;
    const int m_base = blockIdx.x * 64 + wave * 16;
    const int n_base = blockIdx.y * 64;

    const int ar   = m_base + (lane & 15);
    const int koff = (lane >> 4) * 8;    // A K-group offset
    const int kb2  = (lane >> 4) * 8;    // B K-group offset

    const _Float16* Arow = A  + (long)ar * lda + koff;
    const _Float16* B0   = Bw + (long)(n_base + (lane & 15)) * K + kb2;

    v8f acc[4] = {v8f{}, v8f{}, v8f{}, v8f{}};

    for (int kb = 0; kb < K; kb += 32) {
        // ---- load phase: A fragment + all four B fragments ----
        v8h alo = *reinterpret_cast<const v8h*>(Arow + kb);
        v8h ahi = *reinterpret_cast<const v8h*>(Arow + kb + 16);
        v8h blo[4], bhi[4];
#pragma unroll
        for (int sub = 0; sub < 4; ++sub) {
            const _Float16* Br = B0 + (long)sub * 16 * K + kb;
            blo[sub] = *reinterpret_cast<const v8h*>(Br);
            bhi[sub] = *reinterpret_cast<const v8h*>(Br + 16);
        }

        v16h a;
#pragma unroll
        for (int e = 0; e < 8; ++e) { a[e] = alo[e]; a[e + 8] = ahi[e]; }

        // ---- compute phase: 4 WMMAs reusing the A fragment ----
#pragma unroll
        for (int sub = 0; sub < 4; ++sub) {
            v16h b;
#pragma unroll
            for (int e = 0; e < 8; ++e) { b[e] = blo[sub][e]; b[e + 8] = bhi[sub][e]; }
            acc[sub] = __builtin_amdgcn_wmma_f32_16x16x32_f16(
                /*neg_a=*/false, a, /*neg_b=*/false, b,
                /*c_mod=*/(short)0, acc[sub], /*reuse_a=*/false, /*reuse_b=*/false);
        }
    }

    const int m0 = m_base + (lane >> 4) * 8;
#pragma unroll
    for (int sub = 0; sub < 4; ++sub) {
        const int n = n_base + sub * 16 + (lane & 15);
        const float bn = bias ? bias[n] : 0.0f;
#pragma unroll
        for (int g = 0; g < 8; ++g) {
            float v = acc[sub][g] + bn;
            if (do_relu) v = fmaxf(v, 0.0f);
            const long m = m0 + g;
            if (Cf) Cf[m * ldcf + n] = v;
            if (Ch) Ch[m * ldch + n] = (_Float16)v;
        }
    }
}

// ---------------------------------------------------------------------------
// Small helpers
// ---------------------------------------------------------------------------
__global__ void cvt_f16_kernel(const float* __restrict__ s,
                               _Float16* __restrict__ d, int n)
{
    int i = blockIdx.x * blockDim.x + threadIdx.x;
    if (i < n) d[i] = (_Float16)s[i];
}

__global__ void bias_sum_kernel(const float* __restrict__ a,
                                const float* __restrict__ b,
                                float* __restrict__ o, int n)
{
    int i = blockIdx.x * blockDim.x + threadIdx.x;
    if (i < n) o[i] = a[i] + b[i];
}

// ---------------------------------------------------------------------------
// Time encoding + summed relu(code embeddings).  One block per (b,v) pair.
// ---------------------------------------------------------------------------
__global__ void embed_time_kernel(const int* __restrict__ seqs,
                                  const float* __restrict__ tstep,
                                  const float* __restrict__ emb,
                                  const float* __restrict__ tw1,
                                  const float* __restrict__ tb1,
                                  const float* __restrict__ tw2,
                                  const float* __restrict__ tb2,
                                  float* __restrict__ ve32,
                                  _Float16* __restrict__ ve16)
{
    __shared__ float tf[64];
    const int bv = blockIdx.x;       // b*V + v
    const int d  = threadIdx.x;      // 0..255
    const float t = tstep[bv] * (1.0f / 180.0f);
    if (d < 64) {
        float z = t * tw1[d] + tb1[d];
        tf[d] = 1.0f - tanhf(z * z);
    }
    __syncthreads();
    float acc = tb2[d];
#pragma unroll 8
    for (int j = 0; j < 64; ++j) acc += tf[j] * tw2[d * 64 + j];
    const int* row = seqs + bv * KC_;
    for (int k = 0; k < KC_; ++k) {
        int idx = row[k];
        acc += fmaxf(emb[(long)idx * D_ + d], 0.0f);
    }
    ve32[(long)bv * D_ + d] = acc;
    ve16[(long)bv * D_ + d] = (_Float16)acc;
}

// ---------------------------------------------------------------------------
// Masked attention over cached k/v for step i. One wave per (b, head).
// For step==0 the attention output is discarded downstream -> write zeros.
// ---------------------------------------------------------------------------
__global__ void attn_kernel(const float* __restrict__ q,
                            const _Float16* __restrict__ kc,  // [V,B,D]
                            const _Float16* __restrict__ vc,  // [V,B,D]
                            float* __restrict__ ao32,
                            _Float16* __restrict__ ao16,
                            int step)
{
    __shared__ float sc[4][64];
    const int lane = threadIdx.x & 31;
    const int w    = threadIdx.x >> 5;
    const int gw   = blockIdx.x * 4 + w;
    const int b    = gw / NH_;
    const int h    = gw % NH_;
    const int od   = b * D_ + h * HD_ + lane;

    if (step == 0) { ao32[od] = 0.0f; ao16[od] = (_Float16)0.0f; return; }

    const float* qh = q + b * D_ + h * HD_;
    for (int j = lane; j < 64; j += 32) {       // lanes parallel over keys
        float s = 0.0f;
        if (j < step) {
            const _Float16* kr = kc + ((long)j * B_ + b) * D_ + h * HD_;
#pragma unroll
            for (int dd = 0; dd < HD_; ++dd) s += qh[dd] * (float)kr[dd];
            s *= 0.17677669529663687f;          // 1/sqrt(32)
        }
        sc[w][j] = s;
    }
    __syncthreads();

    float mx = -1e30f;
    for (int j = 0; j < step; ++j) mx = fmaxf(mx, sc[w][j]);
    float den = 0.0f, acc = 0.0f;
    for (int j = 0; j < step; ++j) {
        float e = __expf(sc[w][j] - mx);
        den += e;
        acc += e * (float)vc[((long)j * B_ + b) * D_ + h * HD_ + lane];
    }
    float r = acc / den;
    ao32[od] = r;
    ao16[od] = (_Float16)r;
}

// ---------------------------------------------------------------------------
// LayerNorm(e + ao2), select raw e for step 0.  One block per batch row.
// ---------------------------------------------------------------------------
__global__ void ln_select_kernel(const float* __restrict__ ve32,
                                 const float* __restrict__ ao2,
                                 const float* __restrict__ g,
                                 const float* __restrict__ be,
                                 _Float16* __restrict__ ep16,
                                 int step)
{
    __shared__ float red[256];
    const int b = blockIdx.x, d = threadIdx.x;
    const float e = ve32[((long)b * V_ + step) * D_ + d];
    if (step == 0) { ep16[b * D_ + d] = (_Float16)e; return; }
    const float x = e + ao2[b * D_ + d];
    red[d] = x; __syncthreads();
    for (int s = 128; s > 0; s >>= 1) { if (d < s) red[d] += red[d + s]; __syncthreads(); }
    const float mu = red[0] * (1.0f / D_);
    __syncthreads();
    const float c = x - mu;
    red[d] = c * c; __syncthreads();
    for (int s = 128; s > 0; s >>= 1) { if (d < s) red[d] += red[d + s]; __syncthreads(); }
    const float var = red[0] * (1.0f / D_);
    const float y = g[d] * c * rsqrtf(var + 1e-5f) + be[d];
    ep16[b * D_ + d] = (_Float16)y;
}

// ---------------------------------------------------------------------------
// LSTM single step from zero state: c = sig(i)*tanh(g); h = sig(o)*tanh(c)
// ---------------------------------------------------------------------------
__global__ void lstm_kernel(const float* __restrict__ gates, // [B,1024]
                            _Float16* __restrict__ hid16,    // [B,V,H]
                            int step)
{
    const int b = blockIdx.x, d = threadIdx.x;
    const float ig = gates[b * 1024 + d];
    const float gg = gates[b * 1024 + 512 + d];
    const float og = gates[b * 1024 + 768 + d];
    const float c  = (1.0f / (1.0f + __expf(-ig))) * tanhf(gg);
    const float h  = (1.0f / (1.0f + __expf(-og))) * tanhf(c);
    hid16[((long)b * V_ + step) * H_ + d] = (_Float16)h;
}

// ---------------------------------------------------------------------------
// Final 512 -> 2 projection, writes logits straight into d_out.
// ---------------------------------------------------------------------------
__global__ void logits_kernel(const _Float16* __restrict__ z2, // [6400,512]
                              const float* __restrict__ w3,    // [2,512]
                              const float* __restrict__ b3,
                              float* __restrict__ out)
{
    __shared__ float s0[64], s1[64];
    const int r = blockIdx.x, t = threadIdx.x;
    const _Float16* zr = z2 + (long)r * 512;
    float a0 = 0.0f, a1 = 0.0f;
    for (int k = t; k < 512; k += 64) {
        const float zv = (float)zr[k];
        a0 += zv * w3[k];
        a1 += zv * w3[512 + k];
    }
    s0[t] = a0; s1[t] = a1; __syncthreads();
    for (int s = 32; s > 0; s >>= 1) {
        if (t < s) { s0[t] += s0[t + s]; s1[t] += s1[t + s]; }
        __syncthreads();
    }
    if (t == 0) {
        out[r * 2 + 0] = s0[0] + b3[0];
        out[r * 2 + 1] = s1[0] + b3[1];
    }
}

__global__ void zero_kernel(float* __restrict__ p, int n)
{
    int i = blockIdx.x * blockDim.x + threadIdx.x;
    if (i < n) p[i] = 0.0f;
}

__global__ void last_kernel(float* __restrict__ out)
{
    const int t = threadIdx.x;          // 0..255
    const int b = t >> 1, n = t & 1;
    // third output lives after logits (12800) + zeros (12800)
    out[4 * B_ * V_ + t] = out[((long)b * V_ + (V_ - 1)) * 2 + n];
}

// ---------------------------------------------------------------------------
// Host orchestration
// ---------------------------------------------------------------------------
extern "C" void kernel_launch(void* const* d_in, const int* in_sizes, int n_in,
                              void* d_out, int out_size, void* d_ws, size_t ws_size,
                              hipStream_t stream)
{
    (void)in_sizes; (void)n_in; (void)out_size; (void)ws_size;

    const int*   seqs  = (const int*)  d_in[0];
    const float* tstep = (const float*)d_in[1];
    const float* emb   = (const float*)d_in[2];
    const float* tw1   = (const float*)d_in[3];
    const float* tb1   = (const float*)d_in[4];
    const float* tw2   = (const float*)d_in[5];
    const float* tb2   = (const float*)d_in[6];
    const float* Wq    = (const float*)d_in[7];
    const float* bq    = (const float*)d_in[8];
    const float* Wk    = (const float*)d_in[9];
    const float* bk    = (const float*)d_in[10];
    const float* Wv    = (const float*)d_in[11];
    const float* bv    = (const float*)d_in[12];
    const float* Wo    = (const float*)d_in[13];
    const float* bo    = (const float*)d_in[14];
    const float* Wfc   = (const float*)d_in[15];
    const float* ln_g  = (const float*)d_in[16];
    const float* ln_b  = (const float*)d_in[17];
    const float* Wih   = (const float*)d_in[18];
    const float* b_ih  = (const float*)d_in[20];
    const float* b_hh  = (const float*)d_in[21];
    const float* cw1   = (const float*)d_in[22];
    const float* cb1   = (const float*)d_in[23];
    const float* cw2   = (const float*)d_in[24];
    const float* cb2   = (const float*)d_in[25];
    const float* cw3   = (const float*)d_in[26];
    const float* cb3   = (const float*)d_in[27];
    float* out = (float*)d_out;

    // ---- workspace carve-out (256B-aligned slabs) ----
    char* base = (char*)d_ws;
    size_t cur = 0;
    auto alloc = [&](size_t bytes) -> void* {
        void* p = base + cur;
        cur += (bytes + 255) & ~(size_t)255;
        return p;
    };
    _Float16* wq16   = (_Float16*)alloc(65536 * 2);
    _Float16* wk16   = (_Float16*)alloc(65536 * 2);
    _Float16* wv16   = (_Float16*)alloc(65536 * 2);
    _Float16* wo16   = (_Float16*)alloc(65536 * 2);
    _Float16* wfc16  = (_Float16*)alloc(65536 * 2);
    _Float16* wih16  = (_Float16*)alloc(262144 * 2);
    _Float16* cw116  = (_Float16*)alloc(262144 * 2);
    _Float16* cw216  = (_Float16*)alloc(524288 * 2);
    float*    bihh   = (float*)   alloc(1024 * 4);
    float*    ve32   = (float*)   alloc((size_t)B_ * V_ * D_ * 4);
    _Float16* ve16   = (_Float16*)alloc((size_t)B_ * V_ * D_ * 2);
    _Float16* kc16   = (_Float16*)alloc((size_t)V_ * B_ * D_ * 2);
    _Float16* vc16   = (_Float16*)alloc((size_t)V_ * B_ * D_ * 2);
    _Float16* hid16  = (_Float16*)alloc((size_t)B_ * V_ * H_ * 2);
    float*    q32    = (float*)   alloc((size_t)B_ * D_ * 4);
    float*    ao32   = (float*)   alloc((size_t)B_ * D_ * 4);
    _Float16* ao16   = (_Float16*)alloc((size_t)B_ * D_ * 2);
    _Float16* tmp16  = (_Float16*)alloc((size_t)B_ * D_ * 2);
    float*    ao232  = (float*)   alloc((size_t)B_ * D_ * 4);
    _Float16* ep16   = (_Float16*)alloc((size_t)B_ * D_ * 2);
    float*    gates  = (float*)   alloc((size_t)B_ * 1024 * 4);
    _Float16* z1h    = (_Float16*)alloc((size_t)B_ * V_ * 1024 * 2);
    _Float16* z2h    = (_Float16*)alloc((size_t)B_ * V_ * 512 * 2);

    auto cvt = [&](const float* s, _Float16* d, int n) {
        cvt_f16_kernel<<<(n + 255) / 256, 256, 0, stream>>>(s, d, n);
    };
    auto gemm = [&](const _Float16* A, long lda, const _Float16* Bw,
                    const float* bias, float* Cf, long ldcf,
                    _Float16* Ch, long ldch, int M, int N, int K, int relu) {
        dim3 g(M / 64, N / 64), blk(128);
        wmma_gemm_f16<<<g, blk, 0, stream>>>(A, lda, Bw, bias, Cf, ldcf, Ch, ldch, K, relu);
    };

    // ---- weight conversion + fused LSTM bias ----
    cvt(Wq, wq16, 65536);   cvt(Wk, wk16, 65536);  cvt(Wv, wv16, 65536);
    cvt(Wo, wo16, 65536);   cvt(Wfc, wfc16, 65536);
    cvt(Wih, wih16, 262144); cvt(cw1, cw116, 262144); cvt(cw2, cw216, 524288);
    bias_sum_kernel<<<4, 256, 0, stream>>>(b_ih, b_hh, bihh, 1024);

    // ---- visit embeddings + time encoding ----
    embed_time_kernel<<<B_ * V_, 256, 0, stream>>>(seqs, tstep, emb, tw1, tb1,
                                                   tw2, tb2, ve32, ve16);

    // ---- sequential scan over V steps (incremental KV cache) ----
    for (int i = 0; i < V_; ++i) {
        // q = e_i @ Wq^T + bq   (e_i = ve[:, i, :], strided rows)
        gemm(ve16 + (long)i * D_, (long)V_ * D_, wq16, bq,
             q32, D_, nullptr, 0, B_, D_, D_, 0);
        // masked attention over cached k/v
        attn_kernel<<<(B_ * NH_) / 4, 128, 0, stream>>>(q32, kc16, vc16,
                                                        ao32, ao16, i);
        // out_proj then fc
        gemm(ao16, D_, wo16, bo, nullptr, 0, tmp16, D_, B_, D_, D_, 0);
        gemm(tmp16, D_, wfc16, nullptr, ao232, D_, nullptr, 0, B_, D_, D_, 0);
        // residual + LN (raw e for step 0)
        ln_select_kernel<<<B_, 256, 0, stream>>>(ve32, ao232, ln_g, ln_b, ep16, i);
        // LSTM gates, then h
        gemm(ep16, D_, wih16, bihh, gates, 1024, nullptr, 0, B_, 1024, D_, 0);
        lstm_kernel<<<B_, 256, 0, stream>>>(gates, hid16, i);
        // KV cache update: k_i, v_i from h_i  (h_i = hid16[:, i, :], strided)
        gemm(hid16 + (long)i * H_, (long)V_ * H_, wk16, bk,
             nullptr, 0, kc16 + (long)i * B_ * D_, D_, B_, D_, H_, 0);
        gemm(hid16 + (long)i * H_, (long)V_ * H_, wv16, bv,
             nullptr, 0, vc16 + (long)i * B_ * D_, D_, B_, D_, H_, 0);
    }

    // ---- classifier over [B*V, H] = [6400, 256] ----
    gemm(hid16, H_, cw116, cb1, nullptr, 0, z1h, 1024, B_ * V_, 1024, H_, 1);
    gemm(z1h, 1024, cw216, cb2, nullptr, 0, z2h, 512, B_ * V_, 512, 1024, 1);
    logits_kernel<<<B_ * V_, 64, 0, stream>>>(z2h, cw3, cb3, out);

    // ---- outputs 2 (zeros) and 3 (last-step logits) ----
    zero_kernel<<<(B_ * V_ * 2 + 255) / 256, 256, 0, stream>>>(out + B_ * V_ * 2,
                                                               B_ * V_ * 2);
    last_kernel<<<1, 256, 0, stream>>>(out);
}
